// ExpertsFeedForwardNetwork_40785009442949
// MI455X (gfx1250) — matrix operand order repaired and checked by reading, not compiled, not run on hardware
//
#include <hip/hip_runtime.h>
#include <hip/hip_bf16.h>

// Problem dims (match reference): B=2, S=2048 -> T=4096 tokens, H=1024, F=4096, E=8, K_top=2.
#define T_TOK 4096
#define H_DIM 1024
#define F_DIM 4096
#define E_NUM 8
#define K_TOP 2

typedef __bf16 bf16_t;
typedef __attribute__((ext_vector_type(16))) __bf16 bf16x16;
typedef __attribute__((ext_vector_type(8)))  float  v8f;

#if defined(__AMDGCN__) && __has_builtin(__builtin_amdgcn_global_load_async_to_lds_b128)
#define USE_ASYNC_LDS 1
#else
#define USE_ASYNC_LDS 0
#endif

// ---------------------------------------------------------------------------
// Async global -> LDS copy helpers (gfx1250 async path, ASYNCcnt)
// Builtin signature (from clang diagnostic): src is AS1 (global) v4i*,
// dst is AS3 (LDS) v4i*, then imm offset, imm cpol.
// ---------------------------------------------------------------------------
#if USE_ASYNC_LDS
typedef int v4i_t __attribute__((ext_vector_type(4)));
typedef __attribute__((address_space(1))) void  as1_void;
typedef __attribute__((address_space(3))) void  as3_void;
typedef __attribute__((address_space(1))) v4i_t as1_v4i;
typedef __attribute__((address_space(3))) v4i_t as3_v4i;

__device__ __forceinline__ void async_copy16(const bf16_t* g, bf16_t* l) {
  as1_v4i* gp = (as1_v4i*)(as1_void*)(void*)const_cast<bf16_t*>(g);
  as3_v4i* lp = (as3_v4i*)(as3_void*)(void*)l;
  __builtin_amdgcn_global_load_async_to_lds_b128(gp, lp, 0, 0);
}
#endif

template <int N>
__device__ __forceinline__ void wait_asynccnt() {
#if defined(__AMDGCN__) && __has_builtin(__builtin_amdgcn_s_wait_asynccnt)
  __builtin_amdgcn_s_wait_asynccnt(N);
#elif defined(__AMDGCN__)
  asm volatile("s_wait_asynccnt %0" ::"i"(N) : "memory");
#endif
}

// ---------------------------------------------------------------------------
// gelu (tanh approx, jax.nn.gelu default). Prefer hw v_tanh_f32 on gfx1250.
// ---------------------------------------------------------------------------
__device__ __forceinline__ float gelu_tanh(float x) {
  const float z = 0.7978845608028654f * (x + 0.044715f * x * x * x);
#if defined(__AMDGCN__) && __has_builtin(__builtin_amdgcn_tanhf)
  return 0.5f * x * (1.0f + __builtin_amdgcn_tanhf(z));
#else
  return x * (1.0f / (1.0f + __expf(-2.0f * z)));   // 0.5*(1+tanh(z)) == sigmoid(2z)
#endif
}

// ---------------------------------------------------------------------------
// Elementwise f32 -> bf16 convert (for activations x)
// ---------------------------------------------------------------------------
__global__ void f32_to_bf16_kernel(const float* __restrict__ in,
                                   bf16_t* __restrict__ out, int n) {
  int i = blockIdx.x * blockDim.x + threadIdx.x;
  if (i < n) out[i] = (bf16_t)in[i];
}

// ---------------------------------------------------------------------------
// combine[e*T + t] = sum_k probs[k,t] * (experts[k,t] == e)
// ---------------------------------------------------------------------------
__global__ void combine_kernel(const float* __restrict__ probs,
                               const int* __restrict__ experts,
                               float* __restrict__ combine, int T) {
  int t = blockIdx.x * blockDim.x + threadIdx.x;
  if (t >= T) return;
  float c[E_NUM];
#pragma unroll
  for (int e = 0; e < E_NUM; ++e) c[e] = 0.0f;
#pragma unroll
  for (int k = 0; k < K_TOP; ++k) {
    int e = experts[k * T + t];   // guaranteed in [0, E)
    c[e] += probs[k * T + t];
  }
#pragma unroll
  for (int e = 0; e < E_NUM; ++e) combine[e * T + t] = c[e];
}

// ---------------------------------------------------------------------------
// Transpose + convert: W[K,N] f32 (row-major) -> Wt[N,K] bf16 (row-major).
// ---------------------------------------------------------------------------
__global__ void transpose_convert_kernel(const float* __restrict__ W,
                                         bf16_t* __restrict__ Wt,
                                         int K, int N) {
  __shared__ float tile[32][33];
  const int bx = blockIdx.x * 32;  // N offset
  const int by = blockIdx.y * 32;  // K offset
  const int x = threadIdx.x;       // 0..31
  const int y = threadIdx.y;       // 0..7
#pragma unroll
  for (int i = 0; i < 32; i += 8)
    tile[y + i][x] = W[(size_t)(by + y + i) * N + (bx + x)];
  __syncthreads();
#pragma unroll
  for (int i = 0; i < 32; i += 8)
    Wt[(size_t)(bx + y + i) * K + (by + x)] = (bf16_t)tile[x][y + i];
}

// ---------------------------------------------------------------------------
// WMMA tiled GEMM mainloop.
//   A  : [M, Kd] bf16 row-major
//   Bt : [N, Kd] bf16 row-major (B transposed: K contiguous)
// Block: 256 threads = 8 wave32, tile M=128, N=128, stage depth BK=64
// (two 16x16x32 k-steps -> 16 v_wmma per stage).
// Async path: double-buffered LDS fed by global_load_async_to_lds_b128
// (no VGPR staging -> no spills), s_wait_asynccnt for completion.
// ---------------------------------------------------------------------------
#define BK   64
#define LDK  72   // padded LDS row stride in bf16 elems (144 B, 16B-aligned)

__device__ __forceinline__ bf16x16 load_frag_a(const bf16_t* As, int wm, int i,
                                               int lane, int kk) {
  // CDNA5 16-bit A 16x32 layout: lane m = lane&15; lanes>=16 take K-halves 8..15 / 24..31.
  const int m     = lane & 15;
  const int kbase = (lane >> 4) * 8;
  union { uint4 u[2]; bf16x16 v; } tmp;
  const bf16_t* p = As + (size_t)(wm * 32 + i * 16 + m) * LDK + kk;
  tmp.u[0] = *(const uint4*)(p + kbase);        // K = kbase .. kbase+7
  tmp.u[1] = *(const uint4*)(p + 16 + kbase);   // K = 16+kbase .. 16+kbase+7
  return tmp.v;
}

__device__ __forceinline__ bf16x16 load_frag_b(const bf16_t* Bs, int wn, int j,
                                               int lane, int kk) {
  // B 32x16 layout: column n=lane&15; lanes>=16 take K-halves 8..15 / 24..31 (matches A).
  const int n  = lane & 15;
  const int ks = (lane >> 4) * 8;
  union { uint4 u[2]; bf16x16 v; } tmp;
  const bf16_t* p = Bs + (size_t)(wn * 64 + j * 16 + n) * LDK + kk + ks;
  tmp.u[0] = *(const uint4*)(p);
  tmp.u[1] = *(const uint4*)(p + 16);
  return tmp.v;
}

__device__ __forceinline__ void wmma_stage(const bf16_t* As, const bf16_t* Bs,
                                           v8f acc[2][4], int lane, int wm, int wn) {
#pragma unroll
  for (int kk = 0; kk < BK; kk += 32) {
    bf16x16 af[2], bfr[4];
#pragma unroll
    for (int i = 0; i < 2; ++i) af[i] = load_frag_a(As, wm, i, lane, kk);
#pragma unroll
    for (int j = 0; j < 4; ++j) bfr[j] = load_frag_b(Bs, wn, j, lane, kk);
#pragma unroll
    for (int i = 0; i < 2; ++i)
#pragma unroll
      for (int j = 0; j < 4; ++j)
        acc[i][j] = __builtin_amdgcn_wmma_f32_16x16x32_bf16(
            false, af[i], false, bfr[j], (short)0, acc[i][j], false, false);
  }
}

#if USE_ASYNC_LDS
// Issue one stage: 4 async b128 copies for A slice + 4 for B slice per thread.
__device__ __forceinline__ void issue_stage(const bf16_t* A, const bf16_t* Bt,
                                            bf16_t* As, bf16_t* Bs,
                                            int rowA, int rowB, int Kd, int k0,
                                            int ldRow, int ldCol) {
  const bf16_t* ga = A  + (size_t)rowA * Kd + k0 + ldCol;
  const bf16_t* gb = Bt + (size_t)rowB * Kd + k0 + ldCol;
  bf16_t* la = As + (size_t)ldRow * LDK + ldCol;
  bf16_t* lb = Bs + (size_t)ldRow * LDK + ldCol;
#pragma unroll
  for (int q = 0; q < 4; ++q) {
    async_copy16(ga + q * 8, la + q * 8);
    async_copy16(gb + q * 8, lb + q * 8);
  }
}
#endif

__device__ __forceinline__ void gemm_mainloop(
    const bf16_t* __restrict__ A, const bf16_t* __restrict__ Bt,
    bf16_t* As /*[2][128*LDK]*/, bf16_t* Bs /*[2][128*LDK]*/, v8f acc[2][4],
    int blockM, int blockN, int Kd, int tid, int lane, int wm, int wn) {
  const int ldRow = tid >> 1;           // 0..127
  const int ldCol = (tid & 1) * 32;     // 0 or 32 (bf16 elems; 64 B per thread per tile)
  const int rowA  = blockM + ldRow;
  const int rowB  = blockN + ldRow;

#if USE_ASYNC_LDS
  issue_stage(A, Bt, As, Bs, rowA, rowB, Kd, 0, ldRow, ldCol);
  int s = 0;
  for (int k0 = 0; k0 < Kd; k0 += BK, s ^= 1) {
    bf16_t* curA = As + (size_t)s * (128 * LDK);
    bf16_t* curB = Bs + (size_t)s * (128 * LDK);
    if (k0 + BK < Kd) {
      // issue next stage into the other buffer, then wait only for current stage
      issue_stage(A, Bt, As + (size_t)(s ^ 1) * (128 * LDK),
                  Bs + (size_t)(s ^ 1) * (128 * LDK),
                  rowA, rowB, Kd, k0 + BK, ldRow, ldCol);
      wait_asynccnt<8>();   // async loads complete in order: current stage done
    } else {
      wait_asynccnt<0>();
    }
    __syncthreads();                       // all waves' copies visible
    wmma_stage(curA, curB, acc, lane, wm, wn);
    __syncthreads();                       // buffer s safe to overwrite next iter
  }
#else
  // Fallback: synchronous VGPR staging, single buffer.
  for (int k0 = 0; k0 < Kd; k0 += BK) {
    const uint4* ga = (const uint4*)(A  + (size_t)rowA * Kd + k0 + ldCol);
    const uint4* gb = (const uint4*)(Bt + (size_t)rowB * Kd + k0 + ldCol);
#pragma unroll
    for (int q = 0; q < 4; ++q) {
      *(uint4*)(&As[(size_t)ldRow * LDK + ldCol + q * 8]) = ga[q];
      *(uint4*)(&Bs[(size_t)ldRow * LDK + ldCol + q * 8]) = gb[q];
    }
    __syncthreads();
    wmma_stage(As, Bs, acc, lane, wm, wn);
    __syncthreads();
  }
#endif
}

// ---------------------------------------------------------------------------
// GEMM1: h = gelu_tanh(x @ w1 + b1), output bf16 [M=T, N=F]
// ---------------------------------------------------------------------------
__launch_bounds__(256)
__global__ void gemm1_gelu_kernel(const bf16_t* __restrict__ A,
                                  const bf16_t* __restrict__ Bt,
                                  const float* __restrict__ bias,
                                  bf16_t* __restrict__ Hout,
                                  int M, int N, int Kd) {
#if USE_ASYNC_LDS
  __shared__ bf16_t As[2 * 128 * LDK];
  __shared__ bf16_t Bs[2 * 128 * LDK];
#else
  __shared__ bf16_t As[128 * LDK];
  __shared__ bf16_t Bs[128 * LDK];
#endif
  const int tid  = threadIdx.x;
  const int lane = tid & 31;
  const int wave = tid >> 5;
  const int wm = wave >> 1, wn = wave & 1;
  const int blockM = blockIdx.y * 128;
  const int blockN = blockIdx.x * 128;

  v8f acc[2][4];
#pragma unroll
  for (int i = 0; i < 2; ++i)
#pragma unroll
    for (int j = 0; j < 4; ++j)
      acc[i][j] = (v8f){0.f,0.f,0.f,0.f,0.f,0.f,0.f,0.f};

  gemm_mainloop(A, Bt, As, Bs, acc, blockM, blockN, Kd, tid, lane, wm, wn);

  // D layout: lane n = lane&15, rows m = v + 8*(lane>>4)
  const int nl   = lane & 15;
  const int rsel = (lane >> 4) * 8;
#pragma unroll
  for (int j = 0; j < 4; ++j) {
    const int gn = blockN + wn * 64 + j * 16 + nl;
    const float bv = bias[gn];
#pragma unroll
    for (int i = 0; i < 2; ++i) {
#pragma unroll
      for (int v = 0; v < 8; ++v) {
        const int gm = blockM + wm * 32 + i * 16 + rsel + v;
        Hout[(size_t)gm * N + gn] = (bf16_t)gelu_tanh(acc[i][j][v] + bv);
      }
    }
  }
}

// ---------------------------------------------------------------------------
// GEMM2: out (+)= combine[m] * (h @ w2 + b2), output f32 [M=T, N=H]
// ---------------------------------------------------------------------------
__launch_bounds__(256)
__global__ void gemm2_combine_kernel(const bf16_t* __restrict__ A,
                                     const bf16_t* __restrict__ Bt,
                                     const float* __restrict__ bias,
                                     const float* __restrict__ combine,
                                     float* __restrict__ Out,
                                     int M, int N, int Kd, int accumulate) {
#if USE_ASYNC_LDS
  __shared__ bf16_t As[2 * 128 * LDK];
  __shared__ bf16_t Bs[2 * 128 * LDK];
#else
  __shared__ bf16_t As[128 * LDK];
  __shared__ bf16_t Bs[128 * LDK];
#endif
  const int tid  = threadIdx.x;
  const int lane = tid & 31;
  const int wave = tid >> 5;
  const int wm = wave >> 1, wn = wave & 1;
  const int blockM = blockIdx.y * 128;
  const int blockN = blockIdx.x * 128;

  v8f acc[2][4];
#pragma unroll
  for (int i = 0; i < 2; ++i)
#pragma unroll
    for (int j = 0; j < 4; ++j)
      acc[i][j] = (v8f){0.f,0.f,0.f,0.f,0.f,0.f,0.f,0.f};

  gemm_mainloop(A, Bt, As, Bs, acc, blockM, blockN, Kd, tid, lane, wm, wn);

  const int nl   = lane & 15;
  const int rsel = (lane >> 4) * 8;
#pragma unroll
  for (int j = 0; j < 4; ++j) {
    const int gn = blockN + wn * 64 + j * 16 + nl;
    const float bv = bias[gn];
#pragma unroll
    for (int i = 0; i < 2; ++i) {
#pragma unroll
      for (int v = 0; v < 8; ++v) {
        const int gm = blockM + wm * 32 + i * 16 + rsel + v;
        const float cw  = combine[gm];
        const float val = (acc[i][j][v] + bv) * cw;
        const size_t idx = (size_t)gm * N + gn;
        if (accumulate) Out[idx] += val;  // experts serialized on stream
        else            Out[idx]  = val;  // expert 0 initializes (d_out is poisoned)
      }
    }
  }
}

// ---------------------------------------------------------------------------
// kernel_launch
// Workspace layout (needs ~59 MB):
//   [0,   8M)  xb    : x as bf16,           T*H*2    = 8 MB
//   [8M, 16M)  w1bT  : w1[e]^T bf16 [F,H]   F*H*2    = 8 MB  (reused per expert)
//   [16M,24M)  w2bT  : w2[e]^T bf16 [H,F]   H*F*2    = 8 MB  (reused per expert)
//   [24M,56M)  h     : intermediate bf16    T*F*2    = 32 MB (reused per expert)
//   [56M, + )  comb  : combine f32 [E,T]    E*T*4    = 128 KB
// ---------------------------------------------------------------------------
extern "C" void kernel_launch(void* const* d_in, const int* in_sizes, int n_in,
                              void* d_out, int out_size, void* d_ws, size_t ws_size,
                              hipStream_t stream) {
  const float* x       = (const float*)d_in[0];   // [B,S,H]   f32
  const float* probs   = (const float*)d_in[1];   // [K,B,S,1] f32
  const int*   experts = (const int*)  d_in[2];   // [K,B,S,1] i32
  const float* w1      = (const float*)d_in[3];   // [E,H,F]   f32
  const float* b1      = (const float*)d_in[4];   // [E,F]     f32
  const float* w2      = (const float*)d_in[5];   // [E,F,H]   f32
  const float* b2      = (const float*)d_in[6];   // [E,H]     f32
  float* out = (float*)d_out;                     // [B,S,H]   f32

  char* ws = (char*)d_ws;
  bf16_t* xb   = (bf16_t*)(ws);
  bf16_t* w1bT = (bf16_t*)(ws + (size_t)8  * 1024 * 1024);
  bf16_t* w2bT = (bf16_t*)(ws + (size_t)16 * 1024 * 1024);
  bf16_t* hbuf = (bf16_t*)(ws + (size_t)24 * 1024 * 1024);
  float*  comb = (float*) (ws + (size_t)56 * 1024 * 1024);

  // 1) activations -> bf16
  f32_to_bf16_kernel<<<(T_TOK * H_DIM) / 256, 256, 0, stream>>>(x, xb, T_TOK * H_DIM);
  // 2) per-token combine weights
  combine_kernel<<<T_TOK / 256, 256, 0, stream>>>(probs, experts, comb, T_TOK);

  // 3) per-expert FFN (sequential on stream; ws reused deterministically)
  for (int e = 0; e < E_NUM; ++e) {
    transpose_convert_kernel<<<dim3(F_DIM / 32, H_DIM / 32), dim3(32, 8), 0, stream>>>(
        w1 + (size_t)e * H_DIM * F_DIM, w1bT, H_DIM, F_DIM);
    transpose_convert_kernel<<<dim3(H_DIM / 32, F_DIM / 32), dim3(32, 8), 0, stream>>>(
        w2 + (size_t)e * F_DIM * H_DIM, w2bT, F_DIM, H_DIM);

    // h = gelu(x @ w1 + b1)        M=T, N=F, K=H
    gemm1_gelu_kernel<<<dim3(F_DIM / 128, T_TOK / 128), 256, 0, stream>>>(
        xb, w1bT, b1 + (size_t)e * F_DIM, hbuf, T_TOK, F_DIM, H_DIM);

    // out (+)= combine_e * (h @ w2 + b2)   M=T, N=H, K=F
    gemm2_combine_kernel<<<dim3(H_DIM / 128, T_TOK / 128), 256, 0, stream>>>(
        hbuf, w2bT, b2 + (size_t)e * H_DIM, comb + (size_t)e * T_TOK, out,
        T_TOK, H_DIM, F_DIM, e != 0);
  }
}
